// _MultiScaleEdgeConv_22479858828026
// MI455X (gfx1250) — matrix operand order repaired and checked by reading, not compiled, not run on hardware
//
#include <hip/hip_runtime.h>
#include <hip/hip_bf16.h>

typedef float v2f  __attribute__((ext_vector_type(2)));
typedef float v8f  __attribute__((ext_vector_type(8)));
typedef __bf16 v16bf __attribute__((ext_vector_type(16)));
typedef unsigned int u32x4 __attribute__((ext_vector_type(4)));
typedef int s32x4 __attribute__((ext_vector_type(4)));
typedef int s32x8 __attribute__((ext_vector_type(8)));

#define NPTS 4096
#define KSEL 40
#define NEG_INF (-3.4e38f)

#if defined(__has_builtin)
# if __has_builtin(__builtin_amdgcn_tensor_load_to_lds) && __has_builtin(__builtin_amdgcn_s_wait_tensorcnt)
#  define HAVE_TDM 1
# endif
#endif
#ifndef HAVE_TDM
# define HAVE_TDM 0
#endif

// ---------------------------------------------------------------- zero stats
__global__ void zero_kernel(float* stats) {
    stats[threadIdx.x] = 0.0f;   // 512 floats
}

// ---------------------------------------------------------------- KNN (TDM stage -> WMMA f32 16x16x4 gram -> LDS top-40)
// pair(i,j) = 2 x_i·x_j - |x_i|^2 - |x_j|^2 ; top-40 (descending) per row.
__global__ void knn_kernel(const float* __restrict__ x, int* __restrict__ idx_out) {
    extern __shared__ float lds[];                 // [0..12287]=pts[3][4096], then dist[16][4096]
    float* pts  = lds;
    float* dist = lds + 3 * NPTS;
    const int b    = blockIdx.x >> 8;              // 256 row-tiles per batch
    const int row0 = (blockIdx.x & 255) * 16;
    const float* xb = x + (size_t)b * 3 * NPTS;
    const int lane = threadIdx.x & 31;
    const int wave = threadIdx.x >> 5;
    const int half = lane >> 4;
    const int nl   = lane & 15;
    const int kk   = half * 2;                     // K components this half owns

#if HAVE_TDM
    if (wave == 0) {
        // Tensor DMA descriptor: 1-D row of 12288 f32, global xb -> LDS offset 0
        const unsigned long long ga = (unsigned long long)(uintptr_t)xb;
        u32x4 g0;
        g0.x = 1u;                                             // count=1 (user D#)
        g0.y = 0u;                                             // lds_addr = 0
        g0.z = (unsigned int)(ga & 0xffffffffu);               // global_addr[31:0]
        g0.w = (unsigned int)((ga >> 32) & 0x01ffffffu)        // global_addr[56:32]
             | 0x80000000u;                                    // type=2 ("image")
        s32x8 g1;
        g1[0] = 0x00020000;                 // workgroup_mask=0, data_size=4B
        g1[1] = (int)(0x3000u << 16);       // tensor_dim0[15:0]=12288 in bits 63:48
        g1[2] = (int)(1u << 16);            // tensor_dim0[31:16]=0 ; tensor_dim1[15:0]=1
        g1[3] = (int)(0x3000u << 16);       // tensor_dim1[31:16]=0 ; tile_dim0=12288
        g1[4] = 1;                          // tile_dim1=1, tile_dim2=0
        g1[5] = 0x3000;                     // tensor_dim0_stride[31:0]=12288
        g1[6] = 0;                          // stride hi bits
        g1[7] = 0;
        const s32x4 z4 = {0, 0, 0, 0};
#if __clang_major__ >= 23
        const s32x8 z8 = {0, 0, 0, 0, 0, 0, 0, 0};
        __builtin_amdgcn_tensor_load_to_lds(g0, g1, z4, z4, z8, 0);
#else
        __builtin_amdgcn_tensor_load_to_lds(g0, g1, z4, z4, 0);
#endif
        __builtin_amdgcn_s_wait_tensorcnt((short)0);
    }
    __syncthreads();
#else
    for (int i = threadIdx.x; i < 3 * NPTS; i += blockDim.x) pts[i] = xb[i];
    __syncthreads();
#endif

    // A fragment: rows = 16 query points, K = (2x, 2y, 2z, -1)   (reads LDS-resident pts)
    v2f afrag;
    {
        const int m = row0 + nl;
        afrag.x = (kk     < 3) ? 2.0f * pts[kk       * NPTS + m] : -1.0f;
        afrag.y = (kk + 1 < 3) ? 2.0f * pts[(kk + 1) * NPTS + m] : -1.0f;
    }
    // per-lane |x_i|^2 for the 8 C-matrix rows this lane holds (M = r + 8*half)
    float rowxx[8];
#pragma unroll
    for (int r = 0; r < 8; ++r) {
        const int m = row0 + r + half * 8;
        const float px = pts[m], py = pts[NPTS + m], pz = pts[2 * NPTS + m];
        rowxx[r] = px * px + py * py + pz * pz;
    }

    // distance tiles: 8 waves stripe the 256 candidate tiles
    for (int t = wave; t < NPTS / 16; t += 8) {
        const int j0 = t * 16;
        const int j  = j0 + nl;
        const float cx = pts[j], cy = pts[NPTS + j], cz = pts[2 * NPTS + j];
        const float comp[4] = { cx, cy, cz, cx * cx + cy * cy + cz * cz };
        v2f bfrag; bfrag.x = comp[kk]; bfrag.y = comp[kk + 1];
        v8f acc = {0.f,0.f,0.f,0.f,0.f,0.f,0.f,0.f};
        acc = __builtin_amdgcn_wmma_f32_16x16x4_f32(false, afrag, false, bfrag,
                                                    (short)0, acc, false, false);
#pragma unroll
        for (int r = 0; r < 8; ++r)
            dist[(r + half * 8) * NPTS + j0 + nl] = acc[r] - rowxx[r];
    }
    __syncthreads();

    // top-40 per row: persistent per-lane stripe maxima, owner-lane rescan
    for (int rr = 0; rr < 2; ++rr) {
        const int row = wave * 2 + rr;
        float* drow = dist + row * NPTS;
        float bv = NEG_INF; int bi = lane;
        for (int j = lane; j < NPTS; j += 32) {
            const float v = drow[j];
            if (v > bv) { bv = v; bi = j; }
        }
        for (int t = 0; t < KSEL; ++t) {
            float rv = bv; int ri = bi;
#pragma unroll
            for (int off = 16; off; off >>= 1) {
                const float ov = __shfl_xor(rv, off, 32);
                const int   oi = __shfl_xor(ri, off, 32);
                if (ov > rv || (ov == rv && oi < ri)) { rv = ov; ri = oi; }
            }
            if (lane == 0)
                idx_out[((size_t)b * NPTS + row0 + row) * KSEL + t] = ri;
            if ((ri & 31) == lane) {               // owner masks + rescans its stripe
                drow[ri] = NEG_INF;
                bv = NEG_INF; bi = lane;
                for (int j = lane; j < NPTS; j += 32) {
                    const float v = drow[j];
                    if (v > bv) { bv = v; bi = j; }
                }
            }
        }
    }
}

// ---------------------------------------------------------------- edge-feature moments (for analytic BN)
// stats: [0..20]=SS_s(ut), [21..26]=Se_s, [27..47]=SS_l(ut), [48..53]=Se_l
__global__ void edge_stats_kernel(const float* __restrict__ x,
                                  const int* __restrict__ idx,
                                  float* __restrict__ stats) {
    const int p = blockIdx.x * blockDim.x + threadIdx.x;     // 0..B*N-1
    const int b = p >> 12, n = p & (NPTS - 1);
    const float* xb = x + (size_t)b * 3 * NPTS;
    const float ctr[3] = { xb[n], xb[NPTS + n], xb[2 * NPTS + n] };
    const int* row = idx + (size_t)p * KSEL;
    float se_s[6] = {}, se_l[6] = {}, ss_s[21] = {}, ss_l[21] = {};
    for (int kk = 0; kk < KSEL; ++kk) {
        const int j = row[kk];
        float e[6];
        e[0] = xb[j] - ctr[0]; e[1] = xb[NPTS + j] - ctr[1]; e[2] = xb[2 * NPTS + j] - ctr[2];
        e[3] = ctr[0]; e[4] = ctr[1]; e[5] = ctr[2];
        int t = 0;
#pragma unroll
        for (int i = 0; i < 6; ++i) {
            se_l[i] += e[i];
            if (kk < 20) se_s[i] += e[i];
#pragma unroll
            for (int jj = i; jj < 6; ++jj, ++t) {
                const float pr = e[i] * e[jj];
                ss_l[t] += pr;
                if (kk < 20) ss_s[t] += pr;
            }
        }
    }
#pragma unroll
    for (int off = 16; off; off >>= 1) {
#pragma unroll
        for (int i = 0; i < 6; ++i)  { se_s[i] += __shfl_xor(se_s[i], off, 32); se_l[i] += __shfl_xor(se_l[i], off, 32); }
#pragma unroll
        for (int i = 0; i < 21; ++i) { ss_s[i] += __shfl_xor(ss_s[i], off, 32); ss_l[i] += __shfl_xor(ss_l[i], off, 32); }
    }
    if ((threadIdx.x & 31) == 0) {
#pragma unroll
        for (int i = 0; i < 21; ++i) { atomicAdd(stats + i, ss_s[i]); atomicAdd(stats + 27 + i, ss_l[i]); }
#pragma unroll
        for (int i = 0; i < 6; ++i)  { atomicAdd(stats + 21 + i, se_s[i]); atomicAdd(stats + 48 + i, se_l[i]); }
    }
}

// ---------------------------------------------------------------- fold moments into per-channel BN scale/bias
// coef out: [192..255]=scale_s, [256..319]=bias_s, [320..383]=scale_l, [384..447]=bias_l
__device__ __forceinline__ void bn_fold(const float* wo, const float* ss, const float* se,
                                        float gamma, float beta, float cnt,
                                        float* sc_out, float* bi_out) {
    float mu = 0.f;
#pragma unroll
    for (int i = 0; i < 6; ++i) mu += wo[i] * se[i];
    mu /= cnt;
    float ex2 = 0.f; int t = 0;
#pragma unroll
    for (int i = 0; i < 6; ++i)
#pragma unroll
        for (int j = i; j < 6; ++j, ++t) {
            const float c = wo[i] * wo[j] * ss[t];
            ex2 += (i == j) ? c : 2.0f * c;
        }
    ex2 /= cnt;
    const float var = ex2 - mu * mu;
    const float rs = rsqrtf(var + 1e-5f) * gamma;
    *sc_out = rs;
    *bi_out = beta - mu * rs;
}

__global__ void bn_prep_kernel(const float* w_s, const float* g_s, const float* b_s,
                               const float* w_l, const float* g_l, const float* b_l,
                               float* stats) {
    const int o = threadIdx.x;   // 64
    bn_fold(w_s + o * 6, stats +  0, stats + 21, g_s[o], b_s[o],
            8.0f * NPTS * 20.0f, stats + 192 + o, stats + 256 + o);
    bn_fold(w_l + o * 6, stats + 27, stats + 48, g_l[o], b_l[o],
            8.0f * NPTS * 40.0f, stats + 320 + o, stats + 384 + o);
}

// ---------------------------------------------------------------- edge conv: max_k(e·w^T) via WMMA, BN+leaky, bf16 out
__global__ void edge_conv_kernel(const float* __restrict__ x, const int* __restrict__ idx,
                                 const float* __restrict__ w,
                                 const float* __restrict__ bnsc, const float* __restrict__ bnbi,
                                 __bf16* __restrict__ out, int K) {
    const int lane = threadIdx.x & 31, wave = threadIdx.x >> 5;
    const int half = lane >> 4, nl = lane & 15;
    const int tile = blockIdx.x * 8 + wave;        // 2048 point-tiles
    const int b = tile >> 8;
    const int p0 = (tile & 255) * 16;
    const float* xb = x + (size_t)b * 3 * NPTS;
    const int p = p0 + nl;
    const float ctr[3] = { xb[p], xb[NPTS + p], xb[2 * NPTS + p] };

    // B fragments: w[64][6], 4 column tiles x 2 K-steps (K comps 0..3 / 4..7, 6..7 zero)
    v2f bf[4][2];
#pragma unroll
    for (int ct = 0; ct < 4; ++ct) {
        const int o = ct * 16 + nl;
#pragma unroll
        for (int s = 0; s < 2; ++s) {
            const int c0 = s * 4 + half * 2;
            bf[ct][s].x = (c0     < 6) ? w[o * 6 + c0]     : 0.f;
            bf[ct][s].y = (c0 + 1 < 6) ? w[o * 6 + c0 + 1] : 0.f;
        }
    }
    v8f mx[4];
#pragma unroll
    for (int ct = 0; ct < 4; ++ct)
#pragma unroll
        for (int r = 0; r < 8; ++r) mx[ct][r] = NEG_INF;

    const int* irow = idx + ((size_t)b * NPTS + p) * KSEL;
    for (int kk = 0; kk < K; ++kk) {
        __builtin_prefetch(irow + kk + 8, 0, 0);
        const int j = irow[kk];
        float e[8];
        e[0] = xb[j] - ctr[0]; e[1] = xb[NPTS + j] - ctr[1]; e[2] = xb[2 * NPTS + j] - ctr[2];
        e[3] = ctr[0]; e[4] = ctr[1]; e[5] = ctr[2]; e[6] = 0.f; e[7] = 0.f;
        const int c0 = half * 2;
        v2f af0, af1;
        af0.x = e[c0];     af0.y = e[c0 + 1];
        af1.x = e[4 + c0]; af1.y = e[4 + c0 + 1];
#pragma unroll
        for (int ct = 0; ct < 4; ++ct) {
            v8f acc = {0.f,0.f,0.f,0.f,0.f,0.f,0.f,0.f};
            acc = __builtin_amdgcn_wmma_f32_16x16x4_f32(false, af0, false, bf[ct][0],
                                                        (short)0, acc, false, false);
            acc = __builtin_amdgcn_wmma_f32_16x16x4_f32(false, af1, false, bf[ct][1],
                                                        (short)0, acc, false, false);
#pragma unroll
            for (int r = 0; r < 8; ++r) mx[ct][r] = fmaxf(mx[ct][r], acc[r]);
        }
    }
    // BN + leaky (monotone => commutes with max), store bf16 [point][64]
#pragma unroll
    for (int ct = 0; ct < 4; ++ct) {
        const int o = ct * 16 + nl;
        const float sc = bnsc[o], bi = bnbi[o];
#pragma unroll
        for (int r = 0; r < 8; ++r) {
            const int pm = p0 + r + half * 8;
            float v = mx[ct][r] * sc + bi;
            v = (v > 0.f) ? v : 0.2f * v;
            out[((size_t)b * NPTS + pm) * 64 + o] = (__bf16)v;
        }
    }
}

// ---------------------------------------------------------------- fuse GEMM [32768x128]x[128x64] via WMMA bf16 16x16x32
__global__ void fuse_kernel(const __bf16* __restrict__ hs, const __bf16* __restrict__ hl,
                            const float* __restrict__ w_f,
                            float* __restrict__ g, float* __restrict__ sums) {
    __shared__ __bf16 wlds[64 * 128];
    for (int i = threadIdx.x; i < 64 * 128; i += blockDim.x) wlds[i] = (__bf16)w_f[i];
    __syncthreads();
    const unsigned int* wl32 = (const unsigned int*)wlds;

    const int lane = threadIdx.x & 31, wave = threadIdx.x >> 5;
    const int half = lane >> 4, nl = lane & 15;
    const int p0 = (blockIdx.x * 8 + wave) * 16;   // flat point row
    v8f acc[4];
#pragma unroll
    for (int ct = 0; ct < 4; ++ct) acc[ct] = (v8f){0.f,0.f,0.f,0.f,0.f,0.f,0.f,0.f};

    union PK { v16bf v; unsigned int u[8]; };

#pragma unroll
    for (int ks = 0; ks < 4; ++ks) {
        // A fragment: bf16 16x32 layout; K pairs are 4B-aligned -> u32 loads
        PK a;
        const size_t m = (size_t)(p0 + nl) * 64;
#pragma unroll
        for (int r = 0; r < 8; ++r) {
            const int base = ks * 32 + ((r < 4) ? 0 : 16) + half * 8 + 2 * (r & 3);
            const __bf16* src = (base < 64) ? (hs + m + base) : (hl + m + (base - 64));
            a.u[r] = *(const unsigned int*)src;
        }
#pragma unroll
        for (int ct = 0; ct < 4; ++ct) {
            PK bb;
            const int o = ct * 16 + nl;
#pragma unroll
            for (int r = 0; r < 8; ++r) {
                const int kb = ks * 32 + ((r < 4) ? 0 : 16) + half * 8 + 2 * (r & 3);
                bb.u[r] = wl32[o * 64 + (kb >> 1)];
            }
            acc[ct] = __builtin_amdgcn_wmma_f32_16x16x32_bf16(false, a.v, false, bb.v,
                                                              (short)0, acc[ct], false, false);
        }
    }
    // write g + accumulate per-channel moments (sums[0..63]=Σg, [64..127]=Σg²)
#pragma unroll
    for (int ct = 0; ct < 4; ++ct) {
        const int o = ct * 16 + nl;
        float s1 = 0.f, s2 = 0.f;
#pragma unroll
        for (int r = 0; r < 8; ++r) {
            const int pm = p0 + r + half * 8;
            const float v = acc[ct][r];
            g[(size_t)pm * 64 + o] = v;
            s1 += v; s2 += v * v;
        }
        s1 += __shfl_xor(s1, 16, 32);
        s2 += __shfl_xor(s2, 16, 32);
        if (half == 0) { atomicAdd(sums + o, s1); atomicAdd(sums + 64 + o, s2); }
    }
}

// ---------------------------------------------------------------- final BN + leaky + transpose to [B,O,N]
__global__ void final_kernel(const float* __restrict__ g, const float* __restrict__ sums,
                             const float* __restrict__ gf, const float* __restrict__ bfb,
                             float* __restrict__ out) {
    const size_t i = (size_t)blockIdx.x * blockDim.x + threadIdx.x;   // B*64*N
    const int n = (int)(i & (NPTS - 1));
    const int o = (int)((i >> 12) & 63);
    const int b = (int)(i >> 18);
    const float cnt = 8.0f * NPTS;
    const float mu = sums[o] / cnt;
    const float var = sums[64 + o] / cnt - mu * mu;
    const float rs = rsqrtf(var + 1e-5f) * gf[o];
    float v = (g[((size_t)b * NPTS + n) * 64 + o] - mu) * rs + bfb[o];
    out[i] = (v > 0.f) ? v : 0.2f * v;
}

// ---------------------------------------------------------------- launch
extern "C" void kernel_launch(void* const* d_in, const int* in_sizes, int n_in,
                              void* d_out, int out_size, void* d_ws, size_t ws_size,
                              hipStream_t stream) {
    (void)in_sizes; (void)n_in; (void)out_size; (void)ws_size;
    const float* x   = (const float*)d_in[0];
    const float* w_s = (const float*)d_in[1];
    const float* g_s = (const float*)d_in[2];
    const float* b_s = (const float*)d_in[3];
    const float* w_l = (const float*)d_in[4];
    const float* g_l = (const float*)d_in[5];
    const float* b_l = (const float*)d_in[6];
    const float* w_f = (const float*)d_in[7];
    const float* g_f = (const float*)d_in[8];
    const float* b_f = (const float*)d_in[9];
    float* out = (float*)d_out;

    char* ws = (char*)d_ws;
    float*  stats = (float*)ws;                                    // 512 floats
    int*    idx   = (int*)(ws + 4096);                             // 8*4096*40 ints
    __bf16* hs    = (__bf16*)(ws + 4096 + 5242880);                // 8*4096*64 bf16
    __bf16* hl    = (__bf16*)(ws + 4096 + 5242880 + 4194304);
    float*  g     = (float*)(ws + 4096 + 5242880 + 2 * 4194304);   // 8*4096*64 f32

    zero_kernel<<<1, 512, 0, stream>>>(stats);
    knn_kernel<<<2048, 256, (3 + 16) * NPTS * sizeof(float), stream>>>(x, idx);
    edge_stats_kernel<<<128, 256, 0, stream>>>(x, idx, stats);
    bn_prep_kernel<<<1, 64, 0, stream>>>(w_s, g_s, b_s, w_l, g_l, b_l, stats);
    edge_conv_kernel<<<256, 256, 0, stream>>>(x, idx, w_s, stats + 192, stats + 256, hs, 20);
    edge_conv_kernel<<<256, 256, 0, stream>>>(x, idx, w_l, stats + 320, stats + 384, hl, 40);
    fuse_kernel<<<256, 256, 0, stream>>>(hs, hl, w_f, g, stats + 64);
    final_kernel<<<8192, 256, 0, stream>>>(g, stats + 64, g_f, b_f, out);
}
